// HSTUPointwiseAttention_34471407518413
// MI455X (gfx1250) — matrix operand order repaired and checked by reading, not compiled, not run on hardware
//
#include <hip/hip_runtime.h>
#include <stdint.h>

typedef unsigned short u16;
typedef __attribute__((ext_vector_type(16))) __bf16 v16bf;
typedef __attribute__((ext_vector_type(8)))  __bf16 v8bf;
typedef __attribute__((ext_vector_type(8)))  float  v8f;

#define B_  2
#define T_  2048
#define D_  1024
#define H_  16
#define HD_ 64
#define M_  (B_*T_)        // 4096
#define ALPHA_ 0.125f      // 64^-0.5

__device__ inline u16 f2bf(float f) {
  union { float f; unsigned u; } x; x.f = f;
  unsigned u = x.u;
  u += 0x7fffu + ((u >> 16) & 1u);     // round-to-nearest-even
  return (u16)(u >> 16);
}

// silu via v_exp + v_rcp (avoids the IEEE v_div_scale/fixup expansion)
__device__ inline float silu_fast(float a) {
  return a * __builtin_amdgcn_rcpf(1.0f + __expf(-a));
}

// Build a 16x32 bf16 A-fragment for this lane from a row pointer (32 bf16 row).
// Lane holds row (lane&15); elements 0-15 = K[2*kb..2*kb+15] (contiguous, matches B-frags),
// kb = (lane<16 ? 0 : 8).
__device__ inline v16bf make_afrag(const u16* row, int kb) {
  v8bf lo = *(const v8bf*)(row + 2 * kb);
  v8bf hi = *(const v8bf*)(row + 2 * kb + 8);
  v16bf r;
#pragma unroll
  for (int i = 0; i < 8; ++i) { r[i] = lo[i]; r[i + 8] = hi[i]; }
  return r;
}

__device__ inline v8f wmma_bf16(v16bf a, v16bf b, v8f c) {
  return __builtin_amdgcn_wmma_f32_16x16x32_bf16(false, a, false, b, (short)0, c, false, false);
}

// CDNA5 async LDS load: 16B per lane, ASYNCcnt-tracked (ISA 10.x / 15.18.3, GVS mode).
// lds_dst: generic pointer to __shared__ (low 32 bits = LDS byte address).
__device__ inline void async_ld_b128(void* lds_dst, const void* gbase, unsigned byte_off) {
  asm volatile("global_load_async_to_lds_b128 %0, %1, %2"
               :
               : "v"((unsigned)(uintptr_t)lds_dst), "v"(byte_off), "s"(gbase)
               : "memory");
}
__device__ inline void wait_async0() {
  asm volatile("s_wait_asynccnt 0x0" ::: "memory");
}

// ---------------------------------------------------------------------------
// Kernel 1: h = silu(x @ W_in + b_in); split columns into U (f32), V,Q,K (bf16
// head-major [B,H,T,HD]).  M=4096, N=4096, K=1024.  Block tile 128x128, TK=64.
// ---------------------------------------------------------------------------
__launch_bounds__(256)
__global__ void hstu_gemm1(const float* __restrict__ x,
                           const float* __restrict__ Win,
                           const float* __restrict__ bin,
                           float* __restrict__ u,
                           u16* __restrict__ qb, u16* __restrict__ kb_,
                           u16* __restrict__ vb) {
  __shared__ alignas(32) u16 sA[128 * 64];   // [m][k]
  __shared__ alignas(32) u16 sB[128 * 64];   // [n][k]  (col-major => contiguous B-frags)
  const int bm = blockIdx.x * 128, bn = blockIdx.y * 128;
  const int tid = threadIdx.x, wave = tid >> 5, lane = tid & 31;
  const int wm = (wave & 3) * 32, wn = (wave >> 2) * 64;
  const int l15 = lane & 15, lhalf = lane >> 4;
  v8f acc[2][4] = {};

  for (int k0 = 0; k0 < D_; k0 += 64) {
    __syncthreads();
#pragma unroll
    for (int it = 0; it < 8; ++it) {               // A: 128x64 f32 -> bf16
      int idx = tid + it * 256;                    // float4 index, 16 per row
      int m = idx >> 4, kk = (idx & 15) * 4;
      float4 f = *(const float4*)(x + (size_t)(bm + m) * D_ + k0 + kk);
      u16* d = sA + m * 64 + kk;
      d[0] = f2bf(f.x); d[1] = f2bf(f.y); d[2] = f2bf(f.z); d[3] = f2bf(f.w);
    }
#pragma unroll
    for (int it = 0; it < 8; ++it) {               // B: 64x128 -> [n][k]
      int idx = tid + it * 256;                    // 32 float4 per k-row
      int kk = idx >> 5, nn = (idx & 31) * 4;
      float4 f = *(const float4*)(Win + (size_t)(k0 + kk) * (4 * D_) + bn + nn);
      sB[(nn + 0) * 64 + kk] = f2bf(f.x);
      sB[(nn + 1) * 64 + kk] = f2bf(f.y);
      sB[(nn + 2) * 64 + kk] = f2bf(f.z);
      sB[(nn + 3) * 64 + kk] = f2bf(f.w);
    }
    __syncthreads();
#pragma unroll
    for (int c = 0; c < 2; ++c) {
      v16bf af[2], bf[4];
#pragma unroll
      for (int mt = 0; mt < 2; ++mt)
        af[mt] = make_afrag(sA + (wm + mt * 16 + l15) * 64 + c * 32, lhalf * 8);
#pragma unroll
      for (int nt = 0; nt < 4; ++nt)
        bf[nt] = *(const v16bf*)(sB + (wn + nt * 16 + l15) * 64 + c * 32 + lhalf * 16);
#pragma unroll
      for (int mt = 0; mt < 2; ++mt)
#pragma unroll
        for (int nt = 0; nt < 4; ++nt)
          acc[mt][nt] = wmma_bf16(af[mt], bf[nt], acc[mt][nt]);
    }
  }

  // Segment is block-uniform: each 128-wide N tile lies inside one 1024-wide split.
  const int seg = bn >> 10;
  if (seg == 0) {
#pragma unroll
    for (int mt = 0; mt < 2; ++mt)
#pragma unroll
      for (int nt = 0; nt < 4; ++nt)
#pragma unroll
        for (int r = 0; r < 8; ++r) {
          int gm = bm + wm + mt * 16 + r + lhalf * 8;
          int gn = bn + wn + nt * 16 + l15;
          u[(size_t)gm * D_ + (gn & 1023)] = silu_fast(acc[mt][nt][r] + bin[gn]);
        }
  } else {
    u16* dst = (seg == 1) ? vb : (seg == 2) ? qb : kb_;
#pragma unroll
    for (int mt = 0; mt < 2; ++mt)
#pragma unroll
      for (int nt = 0; nt < 4; ++nt)
#pragma unroll
        for (int r = 0; r < 8; ++r) {
          int gm = bm + wm + mt * 16 + r + lhalf * 8;
          int gn = bn + wn + nt * 16 + l15;
          int col = gn & 1023, hh = col >> 6, dd = col & 63;
          int b = gm >> 11, t = gm & 2047;
          size_t o = (((size_t)(b * H_ + hh)) * T_ + t) * HD_ + dd;
          dst[o] = f2bf(silu_fast(acc[mt][nt][r] + bin[gn]));
        }
  }
}

// ---------------------------------------------------------------------------
// Kernel 2: pointwise-SiLU causal attention. One block per (b*H+h, q-tile 64).
// 4 waves, each owns 16 query rows.  AV -> f32 [B,T,D].
// K tile staged with CDNA5 async LDS loads; V staged transposed manually.
// ---------------------------------------------------------------------------
__launch_bounds__(128)
__global__ void hstu_attn(const u16* __restrict__ qb, const u16* __restrict__ kb_,
                          const u16* __restrict__ vb, const float* __restrict__ rab,
                          float* __restrict__ av) {
  __shared__ alignas(32) u16 sK[64 * 64];    // [j][d]
  __shared__ alignas(32) u16 sVT[64 * 64];   // [d][j]  (transposed for B-frags)
  __shared__ alignas(32) u16 sP[4][16 * 64]; // per-wave P strip
  const int bh = blockIdx.x;                 // b*H + h
  const int h = bh & (H_ - 1), b = bh >> 4;
  const int q0 = blockIdx.y * 64;
  const int tid = threadIdx.x, wave = tid >> 5, lane = tid & 31;
  const int l15 = lane & 15, lhalf = lane >> 4;
  const size_t base = (size_t)bh * T_ * HD_;

  v16bf qa[2];
  {
    const u16* qp = qb + base + (size_t)(q0 + wave * 16 + l15) * HD_;
    qa[0] = make_afrag(qp, lhalf * 8);        // d in [0,32)
    qa[1] = make_afrag(qp + 32, lhalf * 8);   // d in [32,64)
  }
  // Loop-invariant per-lane row constants: row index and 1/(i+1)
  float rden[8];
#pragma unroll
  for (int r = 0; r < 8; ++r)
    rden[r] = __builtin_amdgcn_rcpf((float)(q0 + wave * 16 + r + lhalf * 8 + 1));

  v8f accv[4] = {};
  u16* pw = sP[wave];
  const int nkb = blockIdx.y + 1;             // causal: only key blocks <= q block

  for (int kblk = 0; kblk < nkb; ++kblk) {
    const int k0 = kblk * 64;
    __syncthreads();
#pragma unroll
    for (int it = 0; it < 4; ++it) {          // K tile: 512 x b128 async -> LDS
      int idx = tid + it * 128;
      int j = idx >> 3, dd = (idx & 7) * 8;
      async_ld_b128(sK + j * 64 + dd, kb_,
                    (unsigned)((base + (size_t)(k0 + j) * HD_ + dd) * sizeof(u16)));
    }
#pragma unroll
    for (int it = 0; it < 16; ++it) {         // V tile transposed
      int idx = tid + it * 128;               // uint index, 32 per row
      int j = idx >> 5, dd = (idx & 31) * 2;
      unsigned w = *(const unsigned*)(vb + base + (size_t)(k0 + j) * HD_ + dd);
      sVT[dd * 64 + j]       = (u16)w;
      sVT[(dd + 1) * 64 + j] = (u16)(w >> 16);
    }
    wait_async0();
    __syncthreads();

    // S = Q K^T  (wave's 16 rows x 64 keys), epilogue -> bf16 P in LDS
#pragma unroll
    for (int nt = 0; nt < 4; ++nt) {
      v8f s = {};
#pragma unroll
      for (int c = 0; c < 2; ++c) {
        v16bf kf = *(const v16bf*)(sK + (nt * 16 + l15) * 64 + c * 32 + lhalf * 16);
        s = wmma_bf16(qa[c], kf, s);
      }
#pragma unroll
      for (int r = 0; r < 8; ++r) {
        int il = r + lhalf * 8;
        int gi = q0 + wave * 16 + il;
        int gj = k0 + nt * 16 + l15;
        float val = s[r] * ALPHA_ + rab[((size_t)h * T_ + gi) * T_ + gj];
        float out = (gj > gi) ? 0.0f : silu_fast(val) * rden[r];
        pw[il * 64 + nt * 16 + l15] = f2bf(out);
      }
    }
    // AV += P @ V   (wave-private LDS strip; DS ops are in-order per wave)
#pragma unroll
    for (int c = 0; c < 2; ++c) {
      v16bf pa = make_afrag(pw + l15 * 64 + c * 32, lhalf * 8);
#pragma unroll
      for (int dt = 0; dt < 4; ++dt) {
        v16bf vf = *(const v16bf*)(sVT + (dt * 16 + l15) * 64 + c * 32 + lhalf * 16);
        accv[dt] = wmma_bf16(pa, vf, accv[dt]);
      }
    }
  }

#pragma unroll
  for (int dt = 0; dt < 4; ++dt)
#pragma unroll
    for (int r = 0; r < 8; ++r) {
      int gi = q0 + wave * 16 + r + lhalf * 8;
      int dd = dt * 16 + l15;
      av[((size_t)b * T_ + gi) * D_ + h * HD_ + dd] = accv[dt][r];
    }
}

// ---------------------------------------------------------------------------
// Kernel 3: G = (LayerNorm(AV) * U) -> bf16.  One block per row.
// ---------------------------------------------------------------------------
__launch_bounds__(256)
__global__ void hstu_lnmul(const float* __restrict__ av, const float* __restrict__ u,
                           const float* __restrict__ lnw, const float* __restrict__ lnb,
                           u16* __restrict__ g) {
  __shared__ float red[256];
  const int m = blockIdx.x, tid = threadIdx.x;
  const float* row = av + (size_t)m * D_;
  float vals[4], s = 0.f;
#pragma unroll
  for (int i = 0; i < 4; ++i) { vals[i] = row[tid + i * 256]; s += vals[i]; }
  red[tid] = s; __syncthreads();
  for (int o = 128; o > 0; o >>= 1) { if (tid < o) red[tid] += red[tid + o]; __syncthreads(); }
  float mu = red[0] * (1.f / D_);
  __syncthreads();
  float s2 = 0.f;
#pragma unroll
  for (int i = 0; i < 4; ++i) { float d = vals[i] - mu; s2 += d * d; }
  red[tid] = s2; __syncthreads();
  for (int o = 128; o > 0; o >>= 1) { if (tid < o) red[tid] += red[tid + o]; __syncthreads(); }
  float rs = rsqrtf(red[0] * (1.f / D_) + 1e-5f);
#pragma unroll
  for (int i = 0; i < 4; ++i) {
    int c = tid + i * 256;
    float gv = (vals[i] - mu) * rs * lnw[c] + lnb[c];
    g[(size_t)m * D_ + c] = f2bf(gv * u[(size_t)m * D_ + c]);
  }
}

// ---------------------------------------------------------------------------
// Kernel 4: y = G @ W_out + b_out.  M=4096, N=1024, K=1024.  TK=64.
// A tile (already bf16) staged via async LDS loads.
// ---------------------------------------------------------------------------
__launch_bounds__(256)
__global__ void hstu_gemm2(const u16* __restrict__ g, const float* __restrict__ Wout,
                           const float* __restrict__ bout, float* __restrict__ out) {
  __shared__ alignas(32) u16 sA[128 * 64];
  __shared__ alignas(32) u16 sB[128 * 64];
  const int bm = blockIdx.x * 128, bn = blockIdx.y * 128;
  const int tid = threadIdx.x, wave = tid >> 5, lane = tid & 31;
  const int wm = (wave & 3) * 32, wn = (wave >> 2) * 64;
  const int l15 = lane & 15, lhalf = lane >> 4;
  v8f acc[2][4] = {};

  for (int k0 = 0; k0 < D_; k0 += 64) {
    __syncthreads();
#pragma unroll
    for (int it = 0; it < 4; ++it) {             // A: 1024 x b128 async -> LDS
      int idx = tid + it * 256;
      int m = idx >> 3, kk = (idx & 7) * 8;      // 8 uint4 per 64-u16 row
      async_ld_b128(sA + m * 64 + kk, g,
                    (unsigned)(((size_t)(bm + m) * D_ + k0 + kk) * sizeof(u16)));
    }
#pragma unroll
    for (int it = 0; it < 8; ++it) {             // W_out 64x128 -> [n][k]
      int idx = tid + it * 256;
      int kk = idx >> 5, nn = (idx & 31) * 4;
      float4 f = *(const float4*)(Wout + (size_t)(k0 + kk) * D_ + bn + nn);
      sB[(nn + 0) * 64 + kk] = f2bf(f.x);
      sB[(nn + 1) * 64 + kk] = f2bf(f.y);
      sB[(nn + 2) * 64 + kk] = f2bf(f.z);
      sB[(nn + 3) * 64 + kk] = f2bf(f.w);
    }
    wait_async0();
    __syncthreads();
#pragma unroll
    for (int c = 0; c < 2; ++c) {
      v16bf af[2], bf[4];
#pragma unroll
      for (int mt = 0; mt < 2; ++mt)
        af[mt] = make_afrag(sA + (wm + mt * 16 + l15) * 64 + c * 32, lhalf * 8);
#pragma unroll
      for (int nt = 0; nt < 4; ++nt)
        bf[nt] = *(const v16bf*)(sB + (wn + nt * 16 + l15) * 64 + c * 32 + lhalf * 16);
#pragma unroll
      for (int mt = 0; mt < 2; ++mt)
#pragma unroll
        for (int nt = 0; nt < 4; ++nt)
          acc[mt][nt] = wmma_bf16(af[mt], bf[nt], acc[mt][nt]);
    }
  }
#pragma unroll
  for (int mt = 0; mt < 2; ++mt)
#pragma unroll
    for (int nt = 0; nt < 4; ++nt)
#pragma unroll
      for (int r = 0; r < 8; ++r) {
        int gm = bm + wm + mt * 16 + r + lhalf * 8;
        int gn = bn + wn + nt * 16 + l15;
        out[(size_t)gm * D_ + gn] = acc[mt][nt][r] + bout[gn];
      }
}

// ---------------------------------------------------------------------------
extern "C" void kernel_launch(void* const* d_in, const int* in_sizes, int n_in,
                              void* d_out, int out_size, void* d_ws, size_t ws_size,
                              hipStream_t stream) {
  const float* x    = (const float*)d_in[0];
  // d_in[1] = attention_mask (known causal; implemented analytically)
  const float* rab  = (const float*)d_in[2];
  const float* Win  = (const float*)d_in[3];
  const float* bin  = (const float*)d_in[4];
  const float* Wout = (const float*)d_in[5];
  const float* bout = (const float*)d_in[6];
  const float* lnw  = (const float*)d_in[7];
  const float* lnb  = (const float*)d_in[8];
  float* y = (float*)d_out;

  // workspace layout (needs 64 MB)
  char* ws = (char*)d_ws;
  float* u  = (float*)(ws);                         // 16 MB
  u16*   qb = (u16*)(ws + (16u << 20));             //  8 MB
  u16*   kb = (u16*)(ws + (24u << 20));             //  8 MB
  u16*   vb = (u16*)(ws + (32u << 20));             //  8 MB
  float* av = (float*)(ws + (40u << 20));           // 16 MB
  u16*   g  = (u16*)(ws + (56u << 20));             //  8 MB

  hstu_gemm1<<<dim3(M_ / 128, (4 * D_) / 128), 256, 0, stream>>>(x, Win, bin, u, qb, kb, vb);
  hstu_attn <<<dim3(B_ * H_, T_ / 64), 128, 0, stream>>>(qb, kb, vb, rab, av);
  hstu_lnmul<<<M_, 256, 0, stream>>>(av, u, lnw, lnb, g);
  hstu_gemm2<<<dim3(M_ / 128, D_ / 128), 256, 0, stream>>>(g, Wout, bout, y);
}
